// EnhancedRGCN_29867202576402
// MI455X (gfx1250) — matrix operand order repaired and checked by reading, not compiled, not run on hardware
//
#include <hip/hip_runtime.h>

// EnhancedRGCN for MI455X (gfx1250, wave32).
// All dense matmuls routed through v_wmma_f32_16x16x32_f16 (f16 in, f32 acc),
// register-staged double-buffered LDS tiles, b128 LDS traffic.
// Workspace requirement: ~427M floats (~1.71 GB).

typedef __attribute__((ext_vector_type(16))) _Float16 v16h;
typedef __attribute__((ext_vector_type(8)))  float    v8f;

#define NT_ 300000
#define NC_ 100000
#define NM_ 20000

#define TM 128
#define TN 64
#define TK 32

union Frag { v16h v; uint4 q[2]; };

// ------------------------------------------------------------------
// WMMA GEMM: C[M,N] = act( (A .* rowscale?) [M,K] @ W[K,N] + bias? )
// A,W,C fp32 row-major; converted to f16 while staging in LDS.
// Requires N % 64 == 0. M,K arbitrary (guarded / zero-padded).
// ------------------------------------------------------------------
__global__ __launch_bounds__(256)
void k_gemm(const float* __restrict__ A, const float* __restrict__ W,
            const float* __restrict__ bias, const float* __restrict__ rowscale,
            float* __restrict__ C, int M, int K, int N, int act)
{
  __shared__ alignas(16) _Float16 As[2][TM * TK];   // [m][k] row-major
  __shared__ alignas(16) _Float16 Bs[2][TN * TK];   // [n][k] (transposed from W)

  const int tid  = threadIdx.x;
  const int lane = tid & 31;
  const int w    = tid >> 5;          // wave 0..7
  const int wm   = (w & 3) * 32;      // wave M offset inside tile
  const int wn   = (w >> 2) * 32;     // wave N offset inside tile
  const int lh   = lane >> 4;         // lane half (0/1)
  const int lm   = lane & 15;

  const int M0 = blockIdx.x * TM;
  const int N0 = blockIdx.y * TN;

  v8f acc[2][2];
  for (int i = 0; i < 2; ++i)
    for (int j = 0; j < 2; ++j)
      for (int e = 0; e < 8; ++e) acc[i][j][e] = 0.0f;

  // A-tile cooperative load: thread -> row tid>>1, 16 cols at (tid&1)*16
  const int ar = tid >> 1;
  const int ac = (tid & 1) * 16;
  const int gr = M0 + ar;
  const bool rvalid = (gr < M);
  const float rs = (rvalid && rowscale) ? rowscale[gr] : 1.0f;
  const float* Ap = A + (long)(rvalid ? gr : 0) * K;

  // B-tile cooperative load: thread -> col tid&63, 8 k's at (tid>>6)*8
  const int bn = tid & 63;
  const int bk = (tid >> 6) * 8;
  const float* Wp = W + N0 + bn;

  const int nk = (K + TK - 1) / TK;

  float afr[16];     // register-staged A fragment of next tile
  float bfr[8];      // register-staged B fragment of next tile

  auto fetch = [&](int kt) {
    const int k0 = kt * TK;
    if (rvalid && (k0 + TK <= K)) {
      // fast path: unguarded, loads batch & pipeline
      #pragma unroll
      for (int c = 0; c < 16; ++c) afr[c] = Ap[k0 + ac + c];
    } else if (rvalid) {
      #pragma unroll
      for (int c = 0; c < 16; ++c) {
        int k = k0 + ac + c;
        afr[c] = (k < K) ? Ap[k] : 0.0f;
      }
    } else {
      #pragma unroll
      for (int c = 0; c < 16; ++c) afr[c] = 0.0f;
    }
    if (k0 + TK <= K) {
      #pragma unroll
      for (int c = 0; c < 8; ++c) bfr[c] = Wp[(long)(k0 + bk + c) * N];
    } else {
      #pragma unroll
      for (int c = 0; c < 8; ++c) {
        int k = k0 + bk + c;
        bfr[c] = (k < K) ? Wp[(long)k * N] : 0.0f;
      }
    }
  };

  auto stage = [&](int buf) {
    union { _Float16 h[16]; uint4 q[2]; } pa;
    #pragma unroll
    for (int c = 0; c < 16; ++c) pa.h[c] = (_Float16)(afr[c] * rs);
    *(uint4*)&As[buf][ar * TK + ac]     = pa.q[0];   // 16B-aligned b128 stores
    *(uint4*)&As[buf][ar * TK + ac + 8] = pa.q[1];
    union { _Float16 h[8]; uint4 q; } pb;
    #pragma unroll
    for (int c = 0; c < 8; ++c) pb.h[c] = (_Float16)bfr[c];
    *(uint4*)&Bs[buf][bn * TK + bk] = pb.q;
  };

  fetch(0);
  stage(0);
  __syncthreads();

  for (int kt = 0; kt < nk; ++kt) {
    const int cur = kt & 1;

    if (kt + 1 < nk) fetch(kt + 1);          // overlap global fetch with WMMA
    if (kt + 2 < nk && rvalid)               // hint tile-after-next of A
      __builtin_prefetch(Ap + (kt + 2) * TK + ac, 0, 1);

    Frag a[2], b[2];
    #pragma unroll
    for (int i = 0; i < 2; ++i) {
      // A 16x32 f16 frag: lanes 0-15 -> M=lm, K in {0..7,16..23}; lanes 16-31 shift K by 8
      int arow = wm + i * 16 + lm;
      int kk = lh * 8;
      a[i].q[0] = *(const uint4*)&As[cur][arow * TK + kk];        // K kk..kk+7
      a[i].q[1] = *(const uint4*)&As[cur][arow * TK + kk + 16];   // K kk+16..kk+23
    }
    #pragma unroll
    for (int j = 0; j < 2; ++j) {
      // B 32x16 f16 frag: N=lm, K in {kk..kk+7, kk+16..kk+23}, kk = lane half * 8 (matches A)
      int brow = wn + j * 16 + lm;
      int kk = lh * 8;
      b[j].q[0] = *(const uint4*)&Bs[cur][brow * TK + kk];
      b[j].q[1] = *(const uint4*)&Bs[cur][brow * TK + kk + 16];
    }

    #pragma unroll
    for (int i = 0; i < 2; ++i)
      #pragma unroll
      for (int j = 0; j < 2; ++j)
        acc[i][j] = __builtin_amdgcn_wmma_f32_16x16x32_f16(
            false, a[i].v, false, b[j].v, (short)0, acc[i][j], false, false);

    if (kt + 1 < nk) stage(cur ^ 1);         // write next tile into other buffer
    __syncthreads();
  }

  // Epilogue: C/D layout VGPR r -> M = r + lh*8, N = lm
  #pragma unroll
  for (int i = 0; i < 2; ++i)
    #pragma unroll
    for (int j = 0; j < 2; ++j)
      #pragma unroll
      for (int r = 0; r < 8; ++r) {
        int row = M0 + wm + i * 16 + r + lh * 8;
        int col = N0 + wn + j * 16 + lm;
        if (row < M) {
          float v = acc[i][j][r];
          if (bias) v += bias[col];
          if (act)  v = fmaxf(v, 0.0f);
          C[(long)row * N + col] = v;
        }
      }
}

// ------------------------------------------------------------------ helpers
__global__ void k_zero(float* __restrict__ p, long n) {
  long i = (long)blockIdx.x * blockDim.x + threadIdx.x;
  if (i < n) p[i] = 0.0f;
}

__global__ void k_count(const int* __restrict__ ids, int n, float* __restrict__ deg) {
  int i = blockIdx.x * blockDim.x + threadIdx.x;
  if (i < n) atomicAdd(&deg[ids[i]], 1.0f);
}

__global__ void k_rsqrt_clamp(float* __restrict__ p, int n) {
  int i = blockIdx.x * blockDim.x + threadIdx.x;
  if (i < n) p[i] = rsqrtf(fmaxf(p[i], 1.0f));
}

// out[t,f] = act( hc[ic[t],f] + hm[im[t],f] + b0[f] + b1[f] )
__global__ void k_gather(const float* __restrict__ hc, const float* __restrict__ hm,
                         const int* __restrict__ ic, const int* __restrict__ im,
                         const float* __restrict__ b0, const float* __restrict__ b1,
                         float* __restrict__ out, int n, int d, int act) {
  long idx = (long)blockIdx.x * blockDim.x + threadIdx.x;
  if (idx >= (long)n * d) return;
  int t = (int)(idx / d);
  int f = (int)(idx - (long)t * d);
  float v = hc[(long)ic[t] * d + f] + hm[(long)im[t] * d + f] + b0[f] + b1[f];
  out[idx] = act ? fmaxf(v, 0.0f) : v;
}

// agg[dst[t],f] += g[t,f]
__global__ void k_scatter_add(const float* __restrict__ g, const int* __restrict__ dst,
                              float* __restrict__ agg, int n, int d) {
  long idx = (long)blockIdx.x * blockDim.x + threadIdx.x;
  if (idx >= (long)n * d) return;
  int t = (int)(idx / d);
  int f = (int)(idx - (long)t * d);
  atomicAdd(&agg[(long)dst[t] * d + f], g[idx]);
}

// out[c,f] = act( agg[c,f]*rs[c] + b[f] )
__global__ void k_scale_bias_act(const float* __restrict__ agg, const float* __restrict__ rs,
                                 const float* __restrict__ b, float* __restrict__ out,
                                 int n, int d, int act) {
  long idx = (long)blockIdx.x * blockDim.x + threadIdx.x;
  if (idx >= (long)n * d) return;
  int c = (int)(idx / d);
  int f = (int)(idx - (long)c * d);
  float v = agg[idx] * rs[c] + b[f];
  out[idx] = act ? fmaxf(v, 0.0f) : v;
}

// out[i] = dot(A[i,:], w[0:K])
__global__ void k_rowdot(const float* __restrict__ A, const float* __restrict__ w,
                         float* __restrict__ out, int n, int K) {
  int i = blockIdx.x * blockDim.x + threadIdx.x;
  if (i >= n) return;
  const float* a = A + (long)i * K;
  float s = 0.0f;
  for (int f = 0; f < K; ++f) s += a[f] * w[f];
  out[i] = s;
}

// out[i] = dot(A[i,:], w) + b[0]
__global__ void k_rowdot_bias(const float* __restrict__ A, const float* __restrict__ w,
                              const float* __restrict__ b, float* __restrict__ out,
                              int n, int K) {
  int i = blockIdx.x * blockDim.x + threadIdx.x;
  if (i >= n) return;
  const float* a = A + (long)i * K;
  float s = 0.0f;
  for (int f = 0; f < K; ++f) s += a[f] * w[f];
  out[i] = s + b[0];
}

__global__ void k_scatter_scalar(const float* __restrict__ v, const int* __restrict__ dst,
                                 float* __restrict__ agg, int n) {
  int i = blockIdx.x * blockDim.x + threadIdx.x;
  if (i < n) atomicAdd(&agg[dst[i]], v[i]);
}

// out[i] = agg[i]*rs[i] + b[0]
__global__ void k_scale_bias_scalar(const float* __restrict__ agg, const float* __restrict__ rs,
                                    const float* __restrict__ b, float* __restrict__ out, int n) {
  int i = blockIdx.x * blockDim.x + threadIdx.x;
  if (i < n) out[i] = agg[i] * rs[i] + b[0];
}

// ------------------------------------------------------------------ launch
extern "C" void kernel_launch(void* const* d_in, const int* in_sizes, int n_in,
                              void* d_out, int out_size, void* d_ws, size_t ws_size,
                              hipStream_t stream) {
  const float* X      = (const float*)d_in[0];
  const int*   card   = (const int*)  d_in[1];
  const int*   merch  = (const int*)  d_in[2];
  const float* cemb   = (const float*)d_in[3];
  const float* memb   = (const float*)d_in[4];
  const float* pre_Wi = (const float*)d_in[5];
  const float* pre_bi = (const float*)d_in[6];
  const float* pre_Wh = (const float*)d_in[7];
  const float* pre_bh = (const float*)d_in[8];
  const float* pre_Wo = (const float*)d_in[9];
  const float* pre_bo = (const float*)d_in[10];
  const float* c1W    = (const float*)d_in[11];  // (4,128,128)
  const float* c1b    = (const float*)d_in[12];  // (4,128)
  const float* c2W    = (const float*)d_in[13];
  const float* c2b    = (const float*)d_in[14];
  const float* c3Wt   = (const float*)d_in[15];  // (2,128,64)
  const float* c3bt   = (const float*)d_in[16];  // (2,64)
  const float* c3Wo   = (const float*)d_in[17];  // (2,128,1)
  const float* c3bo   = (const float*)d_in[18];  // (2,1)
  const float* po_Wi  = (const float*)d_in[19];
  const float* po_bi  = (const float*)d_in[20];
  const float* po_Wh  = (const float*)d_in[21];
  const float* po_bh  = (const float*)d_in[22];
  const float* po_Wo  = (const float*)d_in[23];
  const float* po_bo  = (const float*)d_in[24];

  float* out   = (float*)d_out;
  float* out_t = out;               // [NT_]
  float* out_c = out + NT_;         // [NC_]
  float* out_m = out + NT_ + NC_;   // [NM_]

  // ---- workspace layout (floats) ----
  float* wsp = (float*)d_ws;
  size_t o = 0;
  auto alloc = [&](size_t n) { float* r = wsp + o; o += n; return r; };
  float* rs_c = alloc(NC_);
  float* rs_m = alloc(NM_);
  float* tf1  = alloc((size_t)NT_ * 256);
  float* tf2  = alloc((size_t)NT_ * 256);
  float* tf   = alloc((size_t)NT_ * 128);
  float* hc   = alloc((size_t)NC_ * 128);   // src-side card messages (reused per layer)
  float* hm   = alloc((size_t)NM_ * 128);
  float* hta  = alloc((size_t)NT_ * 128);   // h1_t
  float* htb  = alloc((size_t)NT_ * 128);   // h2_t
  float* gbuf = alloc((size_t)NT_ * 128);   // target->{card,merchant} messages
  float* aggc = alloc((size_t)NC_ * 128);
  float* aggm = alloc((size_t)NM_ * 128);
  float* hca  = alloc((size_t)NC_ * 128);   // h1_c
  float* hcb  = alloc((size_t)NC_ * 128);   // h2_c
  float* hma  = alloc((size_t)NM_ * 128);   // h1_m
  float* hmb  = alloc((size_t)NM_ * 128);   // h2_m
  float* h3t  = alloc((size_t)NT_ * 64);
  float* pp1  = alloc((size_t)NT_ * 64);
  float* pp2  = alloc((size_t)NT_ * 64);
  float* vsc  = alloc(NT_);
  float* asc  = alloc(NC_);
  float* amc  = alloc(NM_);
  (void)ws_size; (void)n_in; (void)in_sizes; (void)out_size;

  auto ew = [&](long n) { return dim3((unsigned)((n + 255) / 256)); };
  auto gemm = [&](const float* A, const float* W, const float* b, const float* rsc,
                  float* C, int M, int K, int N, int act) {
    dim3 g((M + TM - 1) / TM, N / TN);
    k_gemm<<<g, 256, 0, stream>>>(A, W, b, rsc, C, M, K, N, act);
  };

  // ---- degrees -> rsqrt ----
  k_zero<<<ew(NC_), 256, 0, stream>>>(rs_c, NC_);
  k_zero<<<ew(NM_), 256, 0, stream>>>(rs_m, NM_);
  k_count<<<ew(NT_), 256, 0, stream>>>(card,  NT_, rs_c);
  k_count<<<ew(NT_), 256, 0, stream>>>(merch, NT_, rs_m);
  k_rsqrt_clamp<<<ew(NC_), 256, 0, stream>>>(rs_c, NC_);
  k_rsqrt_clamp<<<ew(NM_), 256, 0, stream>>>(rs_m, NM_);

  // ---- target preprocessing FF: 390 -> 256 -> 256 -> 128 ----
  gemm(X,   pre_Wi, pre_bi, nullptr, tf1, NT_, 390, 256, 1);
  gemm(tf1, pre_Wh, pre_bh, nullptr, tf2, NT_, 256, 256, 1);
  gemm(tf2, pre_Wo, pre_bo, nullptr, tf,  NT_, 256, 128, 0);

  // ---- conv1 ----
  gemm(cemb, c1W + 0 * 16384, nullptr, rs_c, hc, NC_, 128, 128, 0);
  gemm(memb, c1W + 1 * 16384, nullptr, rs_m, hm, NM_, 128, 128, 0);
  k_gather<<<ew((long)NT_ * 128), 256, 0, stream>>>(hc, hm, card, merch,
      c1b + 0, c1b + 128, hta, NT_, 128, 1);                          // h1_t

  gemm(tf, c1W + 2 * 16384, nullptr, nullptr, gbuf, NT_, 128, 128, 0);
  k_zero<<<ew((long)NC_ * 128), 256, 0, stream>>>(aggc, (long)NC_ * 128);
  k_scatter_add<<<ew((long)NT_ * 128), 256, 0, stream>>>(gbuf, card, aggc, NT_, 128);
  k_scale_bias_act<<<ew((long)NC_ * 128), 256, 0, stream>>>(aggc, rs_c, c1b + 2 * 128,
      hca, NC_, 128, 1);                                              // h1_c

  gemm(tf, c1W + 3 * 16384, nullptr, nullptr, gbuf, NT_, 128, 128, 0);
  k_zero<<<ew((long)NM_ * 128), 256, 0, stream>>>(aggm, (long)NM_ * 128);
  k_scatter_add<<<ew((long)NT_ * 128), 256, 0, stream>>>(gbuf, merch, aggm, NT_, 128);
  k_scale_bias_act<<<ew((long)NM_ * 128), 256, 0, stream>>>(aggm, rs_m, c1b + 3 * 128,
      hma, NM_, 128, 1);                                              // h1_m

  // ---- conv2 ----
  gemm(hca, c2W + 0 * 16384, nullptr, rs_c, hc, NC_, 128, 128, 0);
  gemm(hma, c2W + 1 * 16384, nullptr, rs_m, hm, NM_, 128, 128, 0);
  k_gather<<<ew((long)NT_ * 128), 256, 0, stream>>>(hc, hm, card, merch,
      c2b + 0, c2b + 128, htb, NT_, 128, 1);                          // h2_t

  gemm(hta, c2W + 2 * 16384, nullptr, nullptr, gbuf, NT_, 128, 128, 0);
  k_zero<<<ew((long)NC_ * 128), 256, 0, stream>>>(aggc, (long)NC_ * 128);
  k_scatter_add<<<ew((long)NT_ * 128), 256, 0, stream>>>(gbuf, card, aggc, NT_, 128);
  k_scale_bias_act<<<ew((long)NC_ * 128), 256, 0, stream>>>(aggc, rs_c, c2b + 2 * 128,
      hcb, NC_, 128, 1);                                              // h2_c

  gemm(hta, c2W + 3 * 16384, nullptr, nullptr, gbuf, NT_, 128, 128, 0);
  k_zero<<<ew((long)NM_ * 128), 256, 0, stream>>>(aggm, (long)NM_ * 128);
  k_scatter_add<<<ew((long)NT_ * 128), 256, 0, stream>>>(gbuf, merch, aggm, NT_, 128);
  k_scale_bias_act<<<ew((long)NM_ * 128), 256, 0, stream>>>(aggm, rs_m, c2b + 3 * 128,
      hmb, NM_, 128, 1);                                              // h2_m

  // ---- conv3: into targets (D_OUT=64, no relu) ----
  gemm(hcb, c3Wt + 0 * 128 * 64, nullptr, rs_c, hc, NC_, 128, 64, 0);
  gemm(hmb, c3Wt + 1 * 128 * 64, nullptr, rs_m, hm, NM_, 128, 64, 0);
  k_gather<<<ew((long)NT_ * 64), 256, 0, stream>>>(hc, hm, card, merch,
      c3bt + 0, c3bt + 64, h3t, NT_, 64, 0);                          // h3_t

  // ---- conv3: into cards (scalar) ----
  k_rowdot<<<ew(NT_), 256, 0, stream>>>(htb, c3Wo + 0, vsc, NT_, 128);
  k_zero<<<ew(NC_), 256, 0, stream>>>(asc, NC_);
  k_scatter_scalar<<<ew(NT_), 256, 0, stream>>>(vsc, card, asc, NT_);
  k_scale_bias_scalar<<<ew(NC_), 256, 0, stream>>>(asc, rs_c, c3bo + 0, out_c, NC_);

  // ---- conv3: into merchants (scalar) ----
  k_rowdot<<<ew(NT_), 256, 0, stream>>>(htb, c3Wo + 128, vsc, NT_, 128);
  k_zero<<<ew(NM_), 256, 0, stream>>>(amc, NM_);
  k_scatter_scalar<<<ew(NT_), 256, 0, stream>>>(vsc, merch, amc, NT_);
  k_scale_bias_scalar<<<ew(NM_), 256, 0, stream>>>(amc, rs_m, c3bo + 1, out_m, NM_);

  // ---- target postprocessing FF: 64 -> 64 -> 64 -> 1 ----
  gemm(h3t, po_Wi, po_bi, nullptr, pp1, NT_, 64, 64, 1);
  gemm(pp1, po_Wh, po_bh, nullptr, pp2, NT_, 64, 64, 1);
  k_rowdot_bias<<<ew(NT_), 256, 0, stream>>>(pp2, po_Wo, po_bo, out_t, NT_, 64);
}
